// LongcatFlashTopkRouter_43963285242581
// MI455X (gfx1250) — matrix operand order repaired and checked by reading, not compiled, not run on hardware
//
#include <hip/hip_runtime.h>
#include <hip/hip_bf16.h>

typedef float    v2f  __attribute__((ext_vector_type(2)));
typedef float    v4f  __attribute__((ext_vector_type(4)));
typedef float    v8f  __attribute__((ext_vector_type(8)));
typedef _Float16 v4h  __attribute__((ext_vector_type(4)));
typedef _Float16 v16h __attribute__((ext_vector_type(16)));

#define H          2048
#define E_TOT      768
#define TOPK       12
#define M_WG       64          // tokens per workgroup
#define LDS_STRIDE 770         // padded logits row stride (floats)
#define NT_WAVE    12          // 16-wide expert tiles per wave (192 experts)

// ---------------------------------------------------------------------------
// Kernel 1: split W (f32) into f16 hi + f16 residual planes in workspace.
// whi[i] = (f16)W[i]; wlo[i] = (f16)(W[i] - (f32)whi[i])
// ---------------------------------------------------------------------------
__global__ __launch_bounds__(256)
void split_w_kernel(const float* __restrict__ W,
                    _Float16* __restrict__ whi,
                    _Float16* __restrict__ wlo)
{
    const int g = blockIdx.x * blockDim.x + threadIdx.x;   // quad index
    v4f w = *(const v4f*)(W + (size_t)g * 4);
    v4h hi, lo;
    #pragma unroll
    for (int c = 0; c < 4; ++c) {
        const _Float16 h = (_Float16)w[c];
        hi[c] = h;
        lo[c] = (_Float16)(w[c] - (float)h);
    }
    *(v4h*)(whi + (size_t)g * 4) = hi;
    *(v4h*)(wlo + (size_t)g * 4) = lo;
}

// ---------------------------------------------------------------------------
// Kernel 2: fused router GEMM (split-f16 WMMA, f32 accumulate) + softmax/top-12
// ---------------------------------------------------------------------------
__global__ __launch_bounds__(512)
void longcat_router_kernel(const float* __restrict__ hidden,
                           const _Float16* __restrict__ whi,   // [768][2048] f16 hi (wlo at +E_TOT*H)
                           const float* __restrict__ bias,
                           int*   __restrict__ out_idx,
                           float* __restrict__ out_w)
{
    extern __shared__ float lds[];          // [M_WG][LDS_STRIDE] logits

    const int tid  = threadIdx.x;
    const int lane = tid & 31;
    const int wave = tid >> 5;              // 0..15
    const int mw   = wave >> 2;             // 0..3 : 16-row group
    const int nw   = wave &  3;             // 0..3 : 192-expert group
    const int m    = lane & 15;             // A-row / B-col within 16
    const int h    = lane >> 4;             // lane half-group

    // ---- Phase 1: logits = hidden * W^T via 3x v_wmma_f32_16x16x32_f16 per tile
    const int row0 = blockIdx.x * M_WG + mw * 16;
    // A fragment source: 16 consecutive K [h*16, h*16+16) of each K-32 chunk
    const float* ar = hidden + (size_t)(row0 + m) * H + h * 16;
    // B fragment source: 16 consecutive K at h*16 for column expert e; all tiles + lo
    // plane addressed off one base with constant offsets (<2^23 B, fits imm offset)
    const _Float16* wb0 = whi + (size_t)(nw * 192 + m) * H + (size_t)h * 16;

    v8f acc[NT_WAVE] = {};

    for (int kb = 0; kb < H; kb += 32) {
        // load + split A on the fly
        const v8f a0 = *(const v8f*)(ar + kb);        // K = kb + h*16 + [0,8)
        const v8f a1 = *(const v8f*)(ar + kb + 8);    // K = kb + h*16 + 8 + [0,8)
        v16h ahi, alo;
        #pragma unroll
        for (int i = 0; i < 8; ++i) {
            const float f0 = a0[i];
            const _Float16 h0 = (_Float16)f0;
            ahi[i] = h0; alo[i] = (_Float16)(f0 - (float)h0);
            const float f1 = a1[i];
            const _Float16 h1 = (_Float16)f1;
            ahi[i + 8] = h1; alo[i + 8] = (_Float16)(f1 - (float)h1);
        }

        #pragma unroll
        for (int t = 0; t < NT_WAVE; ++t) {
            const v16h bh = *(const v16h*)(wb0 + kb + (size_t)t * 16 * H);
            const v16h bl = *(const v16h*)(wb0 + kb + (size_t)t * 16 * H + (size_t)E_TOT * H);
            acc[t] = __builtin_amdgcn_wmma_f32_16x16x32_f16(
                         false, ahi, false, bh, (short)0, acc[t], false, false);
            acc[t] = __builtin_amdgcn_wmma_f32_16x16x32_f16(
                         false, ahi, false, bl, (short)0, acc[t], false, false);
            acc[t] = __builtin_amdgcn_wmma_f32_16x16x32_f16(
                         false, alo, false, bh, (short)0, acc[t], false, false);
        }
    }

    // D layout: vgpr v, lane l -> row = v + 8*(l>>4), col = l&15
    #pragma unroll
    for (int t = 0; t < NT_WAVE; ++t) {
        const int col = nw * 192 + t * 16 + m;
        #pragma unroll
        for (int v = 0; v < 8; ++v) {
            const int row = mw * 16 + v + 8 * h;
            lds[row * LDS_STRIDE + col] = acc[t][v];
        }
    }
    __syncthreads();

    // ---- Phase 2: softmax + bias-corrected top-12 (4 tokens per wave)
    float bias_r[24];
    #pragma unroll
    for (int j = 0; j < 24; ++j) bias_r[j] = bias[32 * j + lane];

    for (int i = 0; i < 4; ++i) {
        const int tl  = wave * 4 + i;                 // local token 0..63
        const int tok = blockIdx.x * M_WG + tl;       // global token
        const float* Lr = lds + tl * LDS_STRIDE;

        float sc[24];
        #pragma unroll
        for (int j = 0; j < 24; ++j) sc[j] = Lr[32 * j + lane];

        float mx = sc[0];
        #pragma unroll
        for (int j = 1; j < 24; ++j) mx = fmaxf(mx, sc[j]);
        #pragma unroll
        for (int off = 16; off >= 1; off >>= 1)
            mx = fmaxf(mx, __shfl_xor(mx, off, 32));

        float s = 0.0f;
        #pragma unroll
        for (int j = 0; j < 24; ++j) { sc[j] = __expf(sc[j] - mx); s += sc[j]; }
        #pragma unroll
        for (int off = 16; off >= 1; off >>= 1)
            s += __shfl_xor(s, off, 32);

        const float inv = 1.0f / s;
        float corr[24];
        #pragma unroll
        for (int j = 0; j < 24; ++j) {
            sc[j]  *= inv;                  // uncorrected softmax score
            corr[j] = sc[j] + bias_r[j];    // selection key
        }

        float top_s[TOPK];
        int   top_i[TOPK];
        float ssum = 0.0f;

        #pragma unroll
        for (int k = 0; k < TOPK; ++k) {
            float bv = corr[0]; int bi = lane; float bs = sc[0];
            #pragma unroll
            for (int j = 1; j < 24; ++j) {
                const int  e    = 32 * j + lane;
                const bool take = (corr[j] > bv);
                bv = take ? corr[j] : bv;
                bs = take ? sc[j]   : bs;
                bi = take ? e       : bi;
            }
            #pragma unroll
            for (int off = 16; off >= 1; off >>= 1) {
                const float ov = __shfl_xor(bv, off, 32);
                const int   oi = __shfl_xor(bi, off, 32);
                const float os = __shfl_xor(bs, off, 32);
                const bool take = (ov > bv) || (ov == bv && oi < bi);
                bv = take ? ov : bv;
                bi = take ? oi : bi;
                bs = take ? os : bs;
            }
            top_s[k] = bs;
            top_i[k] = bi;
            ssum += bs;
            #pragma unroll
            for (int j = 0; j < 24; ++j)
                corr[j] = (32 * j + lane == bi) ? -1e30f : corr[j];
        }

        if (lane == 0) {
            const float factor = 2.5f / (ssum + 1e-20f);
            #pragma unroll
            for (int k = 0; k < TOPK; ++k) {
                out_idx[(size_t)tok * TOPK + k] = top_i[k];
                out_w  [(size_t)tok * TOPK + k] = top_s[k] * factor;
            }
        }
    }
}

extern "C" void kernel_launch(void* const* d_in, const int* in_sizes, int n_in,
                              void* d_out, int out_size, void* d_ws, size_t ws_size,
                              hipStream_t stream) {
    const float* hidden = (const float*)d_in[0];   // [NTOK, 2048] f32
    const float* Wm     = (const float*)d_in[1];   // [768, 2048]  f32
    const float* bias   = (const float*)d_in[2];   // [768]        f32

    const int ntok = in_sizes[0] / H;              // 32768

    // workspace: f16 hi plane then f16 lo plane of W (6.3 MB total)
    _Float16* whi = (_Float16*)d_ws;
    _Float16* wlo = whi + (size_t)E_TOT * H;

    int*   out_idx = (int*)d_out;                              // [NTOK, 12] int32
    float* out_w   = (float*)d_out + (size_t)ntok * TOPK;      // [NTOK, 12] f32

    // split W once (1.57M elements)
    {
        const int quads = E_TOT * H / 4;
        split_w_kernel<<<dim3(quads / 256), dim3(256), 0, stream>>>(Wm, whi, wlo);
    }

    const dim3 grid(ntok / M_WG);
    const dim3 block(512);
    const size_t shmem = (size_t)M_WG * LDS_STRIDE * sizeof(float);
    longcat_router_kernel<<<grid, block, shmem, stream>>>(hidden, whi, bias, out_idx, out_w);
}
